// HdcGenericEncoder_17222818857018
// MI455X (gfx1250) — matrix operand order repaired and checked by reading, not compiled, not run on hardware
//
#include <hip/hip_runtime.h>

typedef __attribute__((ext_vector_type(16))) _Float16 v16h;
typedef __attribute__((ext_vector_type(8)))  float    v8f;

#define D_DIM 10000
#define T_DIM 2048
#define KPAD  416      // 400 (=4ch*100lev) padded to 13*32 for K-loop
#define LHS   34       // LDS row stride (32 window cols + pad)
#define NCHUNK 128     // 2048 / 16 t-rows
#define NQ     4       // t-dimension split (32 chunks per quarter)

// ---------------------------------------------------------------------------
// PT[d][j] = keys[c,d]*level[l,d] for j=100c+l (exact +-1 in f16), 0 for j>=400
// ---------------------------------------------------------------------------
__global__ void build_PT(const float* __restrict__ keys,
                         const float* __restrict__ level,
                         _Float16* __restrict__ PT) {
  int id = blockIdx.x * 256 + threadIdx.x;
  if (id >= D_DIM * KPAD) return;
  int d = id / KPAD;
  int j = id - d * KPAD;
  float v = 0.f;
  if (j < 400) {
    int c = j / 100;
    int l = j - c * 100;
    v = keys[c * D_DIM + d] * level[l * D_DIM + d];
  }
  PT[id] = (_Float16)v;
}

// ---------------------------------------------------------------------------
// S stored directly in A-fragment order (this code's contiguous k order, same as B):
//   lane m   (0..15): pos 0..15 = K 0..15
//   lane m+16       : pos 0..15 = K 16..31
// element (t, j): chunk=t/16, M=t%16, ks=j/32, kw=j%32
//   fragment half index = ((chunk*13+ks)*32 + lane)*16 + pos
// ---------------------------------------------------------------------------
__global__ void build_S(const float* __restrict__ input,
                        _Float16* __restrict__ Sf) {
  int id = blockIdx.x * 256 + threadIdx.x;
  if (id >= T_DIM * KPAD) return;
  int t = id / KPAD;
  int j = id - t * KPAD;
  float v = 0.f;
  if (j < 400) {
    int c = j / 100;
    int l = j - c * 100;
    int idx = (int)rintf(input[t * 4 + c] * 99.f);   // RNE matches jnp.round
    idx = idx < 0 ? 0 : (idx > 99 ? 99 : idx);
    v = (idx == l) ? 1.f : 0.f;
  }
  int chunk = t >> 4, M = t & 15;
  int ks = j >> 5, kw = j & 31;
  int grp = kw >> 4;
  int lane = M + (grp << 4);
  int pos = kw & 15;
  Sf[(((chunk * 13 + ks) << 5) + lane) * 16 + pos] = (_Float16)v;
}

// ---------------------------------------------------------------------------
// factor[d] = h3[0]*(h6+h3[7])*(h3[1]+h3[2]+h3[3])*(h3[4]+h3[5]+h3[6])
// ---------------------------------------------------------------------------
__global__ void build_factor(const float* __restrict__ feat,
                             const float* __restrict__ W3,
                             const float* __restrict__ b3,
                             const float* __restrict__ W6,
                             const float* __restrict__ b6,
                             float* __restrict__ factor) {
  int d = blockIdx.x * 256 + threadIdx.x;
  if (d >= D_DIM) return;
  const int starts[8] = {0, 9, 12, 15, 18, 21, 24, 27};
  float h[8];
#pragma unroll
  for (int k = 0; k < 8; ++k) {
    const float* w = W3 + ((long)k * D_DIM + d) * 3;
    int s = starts[k];
    float p = w[0] * feat[s] + w[1] * feat[s + 1] + w[2] * feat[s + 2];
    h[k] = cosf(p + b3[k * D_DIM + d]) * sinf(p);
  }
  float p6 = 0.f;
#pragma unroll
  for (int i = 0; i < 6; ++i) p6 += W6[d * 6 + i] * feat[3 + i];
  float h6 = cosf(p6 + b6[d]) * sinf(p6);
  factor[d] = h[0] * (h6 + h[7]) * (h[1] + h[2] + h[3]) * (h[4] + h[5] + h[6]);
}

// ---------------------------------------------------------------------------
// WMMA tile: hv rows [16*chunk, 16*chunk+16), window cols [0,32)
// ---------------------------------------------------------------------------
__device__ __forceinline__ void mma_tile(const v16h* __restrict__ Af,
                                         const _Float16* __restrict__ bp0,
                                         const _Float16* __restrict__ bp1,
                                         int lane, int chunk,
                                         v8f& c0, v8f& c1) {
  const int abase = chunk * 13 * 32 + lane;
  __builtin_prefetch(&Af[abase + 13 * 32], 0, 1);    // next chunk's A frags
#pragma unroll
  for (int ks = 0; ks < 13; ++ks) {
    v16h a  = Af[abase + ks * 32];                   // one aligned 32B load
    v16h b0 = *(const v16h*)(bp0 + ks * 32);
    v16h b1 = *(const v16h*)(bp1 + ks * 32);
    c0 = __builtin_amdgcn_wmma_f32_16x16x32_f16(false, a, false, b0,
                                                (short)0, c0, false, false);
    c1 = __builtin_amdgcn_wmma_f32_16x16x32_f16(false, a, false, b1,
                                                (short)0, c1, false, false);
  }
}

// C layout: VGPR r -> M=r (lanes0-15, N=lane) / M=r+8 (lanes16-31, N=lane-16)
__device__ __forceinline__ void store_tile(float* __restrict__ Lw, int lane,
                                           const v8f& c0, const v8f& c1) {
  const int n = lane & 15;
  const int mb = (lane & 16) ? 8 : 0;
#pragma unroll
  for (int r = 0; r < 8; ++r) {
    Lw[(2 + mb + r) * LHS + n] = c0[r];
    Lw[(2 + mb + r) * LHS + 16 + n] = c1[r];
  }
}

// ---------------------------------------------------------------------------
// One single-wave workgroup per (slot, quarter): slot -> 30 output columns,
// quarter -> 32 of the 128 t-chunks. Rolling 2-row halo in LDS; quarter q>0
// seeds the halo with one extra WMMA tile (chunk 32q-1). Partials are exact
// integers -> order-independent split.
// ---------------------------------------------------------------------------
__global__ void __launch_bounds__(32)
hv_wmma_reduce(const _Float16* __restrict__ S,
               const _Float16* __restrict__ PT,
               float* __restrict__ partial) {
  const int lane = threadIdx.x;
  const int slot = blockIdx.x >> 2;        // 0..333
  const int q    = blockIdx.x & 3;         // 0..3
  const int base = slot * 30;              // first output column
  const int cb = base - 2;                 // window col 0 -> global d (mod D)

  __shared__ float Lw[18 * LHS];

  // Per-lane fixed B column pointers (columns of hv window, wrapped mod D)
  int w0 = cb + (lane & 15);
  int w1 = w0 + 16;
  int d0 = w0 < 0 ? w0 + D_DIM : (w0 >= D_DIM ? w0 - D_DIM : w0);
  int d1 = w1 < 0 ? w1 + D_DIM : (w1 >= D_DIM ? w1 - D_DIM : w1);
  const int kbB = (lane & 16) ? 16 : 0;    // B: lanes16-31 hold K 16..31
  const _Float16* bp0 = PT + (long)d0 * KPAD + kbB;
  const _Float16* bp1 = PT + (long)d1 * KPAD + kbB;
  const v16h* Af = (const v16h*)S;

  const int cfirst = q * (NCHUNK / NQ);
  const int clast  = cfirst + (NCHUNK / NQ);

  // Seed halo rows 16,17 (hv rows 16*cfirst-2, -1)
  if (q == 0) {
    Lw[16 * LHS + lane] = 0.f;
    Lw[17 * LHS + lane] = 0.f;
  } else {
    v8f s0 = {}, s1 = {};
    mma_tile(Af, bp0, bp1, lane, cfirst - 1, s0, s1);
    store_tile(Lw, lane, s0, s1);
  }

  float acc = 0.f;
  for (int chunk = cfirst; chunk < clast; ++chunk) {
    __syncthreads();                       // seed/prev store+product ordered
    // rotate: window rows 16,17 (hv rows t0-2,t0-1) -> rows 0,1
    float r16 = Lw[16 * LHS + lane];
    float r17 = Lw[17 * LHS + lane];
    Lw[0 * LHS + lane] = r16;
    Lw[1 * LHS + lane] = r17;

    v8f cA = {}, cB = {};
    mma_tile(Af, bp0, bp1, lane, chunk, cA, cB);

    __syncthreads();                       // halo reads done before overwrite
    store_tile(Lw, lane, cA, cB);
    __syncthreads();                       // hv tile visible

    // rolled triple product: t = 16*chunk - 2 + r, r = 0..15
    if (lane < 30) {
#pragma unroll
      for (int r = 0; r < 16; ++r)
        acc += Lw[r * LHS + lane] * Lw[(r + 1) * LHS + lane + 1] *
               Lw[(r + 2) * LHS + lane + 2];
    }
  }

  int d = base + lane;
  if (lane < 30 && d < D_DIM) partial[q * D_DIM + d] = acc;
}

// ---------------------------------------------------------------------------
__global__ void finalize(const float* __restrict__ partial,
                         const float* __restrict__ factor,
                         float* __restrict__ out) {
  int d = blockIdx.x * 256 + threadIdx.x;
  if (d >= D_DIM) return;
  float s = partial[d] + partial[D_DIM + d] + partial[2 * D_DIM + d] +
            partial[3 * D_DIM + d];
  float v = s * factor[d];
  out[d] = v > 0.f ? 1.f : -1.f;
}

// ---------------------------------------------------------------------------
extern "C" void kernel_launch(void* const* d_in, const int* in_sizes, int n_in,
                              void* d_out, int out_size, void* d_ws,
                              size_t ws_size, hipStream_t stream) {
  (void)in_sizes; (void)n_in; (void)out_size; (void)ws_size;
  const float* input = (const float*)d_in[0];
  const float* feat  = (const float*)d_in[1];
  const float* keys  = (const float*)d_in[2];
  const float* level = (const float*)d_in[3];
  const float* W3    = (const float*)d_in[4];
  const float* b3    = (const float*)d_in[5];
  const float* W6    = (const float*)d_in[6];
  const float* b6    = (const float*)d_in[7];

  char* ws = (char*)d_ws;
  _Float16* PT  = (_Float16*)ws;                            // 8,320,000 B
  _Float16* S   = (_Float16*)(ws + 8320000);                // 1,703,936 B
  float* factor = (float*)(ws + 8320000 + 1703936);         //    40,000 B
  float* partial = (float*)(ws + 8320000 + 1703936 + 40000);//   160,000 B

  build_PT<<<(D_DIM * KPAD + 255) / 256, 256, 0, stream>>>(keys, level, PT);
  build_S<<<(T_DIM * KPAD + 255) / 256, 256, 0, stream>>>(input, S);
  build_factor<<<(D_DIM + 255) / 256, 256, 0, stream>>>(feat, W3, b3, W6, b6,
                                                        factor);
  // 334 slots x 4 t-quarters, one wave32 each
  hv_wmma_reduce<<<334 * NQ, 32, 0, stream>>>(S, PT, partial);
  finalize<<<(D_DIM + 255) / 256, 256, 0, stream>>>(partial, factor,
                                                    (float*)d_out);
}
